// NTXentLoss_6485400617463
// MI455X (gfx1250) — matrix operand order repaired and checked by reading, not compiled, not run on hardware
//
#include <hip/hip_runtime.h>
#include <hip/hip_bf16.h>

typedef __attribute__((ext_vector_type(16))) _Float16 v16h;
typedef __attribute__((ext_vector_type(8)))  _Float16 v8h;
typedef __attribute__((ext_vector_type(4)))  _Float16 v4h;
typedef __attribute__((ext_vector_type(8)))  float    v8f;

#define B_ROWS   8192
#define N_ROWS   16384           // 2B
#define D_DIM    128
#define N_TILES  (N_ROWS / 16)   // 1024 column tiles of 16
#define WAVES_PER_WG 4
#define ROWT     2               // 16-row tiles per wave (A-reuse)

// ---------------------------------------------------------------------------
// Kernel 1: row L2-normalize z_i / z_j (f32) -> unit-row reps (f16) in d_ws.
// ---------------------------------------------------------------------------
__global__ __launch_bounds__(256) void ntx_normalize_kernel(
    const float* __restrict__ zi, const float* __restrict__ zj,
    _Float16* __restrict__ rep) {
  int row  = blockIdx.x * 8 + (threadIdx.x >> 5);
  int lane = threadIdx.x & 31;
  const float* src = (row < B_ROWS) ? (zi + (size_t)row * D_DIM)
                                    : (zj + (size_t)(row - B_ROWS) * D_DIM);
  float4 v = ((const float4*)src)[lane];
  float ss = v.x * v.x + v.y * v.y + v.z * v.z + v.w * v.w;
  #pragma unroll
  for (int off = 16; off > 0; off >>= 1) ss += __shfl_xor(ss, off, 32);
  float scale = 1.0f / fmaxf(sqrtf(ss), 1e-12f);
  v4h h;
  h[0] = (_Float16)(v.x * scale);
  h[1] = (_Float16)(v.y * scale);
  h[2] = (_Float16)(v.z * scale);
  h[3] = (_Float16)(v.w * scale);
  *(v4h*)(rep + (size_t)row * D_DIM + lane * 4) = h;
}

__global__ void ntx_zero_kernel(float* out) { out[0] = 0.0f; }

// ---------------------------------------------------------------------------
// Kernel 3: fused sim + streaming logsumexp + positive gather.
// Each wave owns TWO 16-row tiles (A in regs, 2x reuse of every B fragment);
// 4 waves/WG share each 16x128 f16 column tile, double-buffered in LDS and
// filled with GLOBAL_LOAD_ASYNC_TO_LDS_B128 (ASYNCcnt) so the copy latency
// hides under the 8 chained V_WMMA_F32_16X16X32_F16 per iteration.
// Diagonal/positive masking runs only on the (wave-uniform) 4/1024 tiles
// that can intersect this wave's rows or their positive block (r ^ 8192).
// ---------------------------------------------------------------------------
__global__ __launch_bounds__(32 * WAVES_PER_WG) void ntx_main_kernel(
    const _Float16* __restrict__ rep, float* __restrict__ out) {
  __shared__ __align__(16) _Float16 tile[2][16 * D_DIM];  // 2 x 4 KB

  const int tid  = threadIdx.x;
  const int lane = tid & 31;
  const int wave = tid >> 5;
  const int ln   = lane & 15;          // N index / A-row M within tile
  const int hi   = lane >> 4;          // half-wave K split
  const int rowBase = (blockIdx.x * WAVES_PER_WG + wave) * (16 * ROWT);
  const int posBase = rowBase ^ B_ROWS;   // positive rows are a 32-row block

  // --- A fragments: 2 row tiles x 4 K-chunks (contiguous k = 16*hi + s, same order as B) ------
  v16h A[ROWT][4];
  #pragma unroll
  for (int rt = 0; rt < ROWT; ++rt) {
    const _Float16* arow = rep + (size_t)(rowBase + rt * 16 + ln) * D_DIM;
    #pragma unroll
    for (int kc = 0; kc < 4; ++kc) {
      v8h l0 = *(const v8h*)(arow + kc * 32 + hi * 16);
      v8h l1 = *(const v8h*)(arow + kc * 32 + hi * 16 + 8);
      #pragma unroll
      for (int i = 0; i < 8; ++i) { A[rt][kc][i] = l0[i]; A[rt][kc][i + 8] = l1[i]; }
    }
  }

  float accS[ROWT][8], posS[ROWT][8];
  #pragma unroll
  for (int rt = 0; rt < ROWT; ++rt)
    #pragma unroll
    for (int v = 0; v < 8; ++v) { accS[rt][v] = 0.0f; posS[rt][v] = 0.0f; }

  // Per-thread async stage of one 4 KB column tile: 2 x b128 per thread.
  auto issue_tile = [&](int buf, int ct) {
    unsigned long long g =
        (unsigned long long)(uintptr_t)(rep + (size_t)ct * 16 * D_DIM) + tid * 16;
    unsigned d = (unsigned)(uintptr_t)(&tile[buf][0]) + tid * 16;
    asm volatile("global_load_async_to_lds_b128 %0, %1, off"
                 :: "v"(d), "v"(g) : "memory");
    asm volatile("global_load_async_to_lds_b128 %0, %1, off"
                 :: "v"((unsigned)(d + 2048)), "v"(g + 2048) : "memory");
  };

  issue_tile(0, 0);                    // prologue: tile 0 in flight

  const float L2E = 1.4426950408889634f;
  const float K1  = 2.0f * L2E;        // exp((s-2)) base-2, s = 2*dot

  for (int ct = 0; ct < N_TILES; ++ct) {
    const int cur = ct & 1, nxt = cur ^ 1;
    __syncthreads();                   // all waves done reading buf `nxt`
    issue_tile(nxt, (ct + 1 < N_TILES) ? ct + 1 : 0);  // overlap with compute
    asm volatile("s_wait_asynccnt 0x2" ::: "memory");  // `cur` loads landed
    __syncthreads();                   // every wave's slice of `cur` visible

    v8f c0 = {}, c1 = {};
    #pragma unroll
    for (int kc = 0; kc < 4; ++kc) {
      v16h b = *(const v16h*)(&tile[cur][ln * D_DIM + kc * 32 + hi * 16]);
      c0 = __builtin_amdgcn_wmma_f32_16x16x32_f16(false, A[0][kc], false, b,
                                                  (short)0, c0, false, false);
      c1 = __builtin_amdgcn_wmma_f32_16x16x32_f16(false, A[1][kc], false, b,
                                                  (short)0, c1, false, false);
    }

    const int colBase = ct * 16;
    // Wave-uniform: does this column tile touch our diagonal or positives?
    const bool special = ((unsigned)(colBase - rowBase) < 32u) ||
                         ((unsigned)(colBase - posBase) < 32u);

    if (__builtin_expect(special, 0)) {
      // D layout: element (VGPR v, lane) = sim[M,N], M = v + 8*hi (+16 c1),
      // N = ln. Positive index of row r is r ^ 8192.
      const int col = colBase + ln;
      const int d   = col - (rowBase + hi * 8);
      const int dp  = (col ^ B_ROWS) - (rowBase + hi * 8);
      #pragma unroll
      for (int v = 0; v < 8; ++v) {
        float s0 = c0[v] * 2.0f;
        float s1 = c1[v] * 2.0f;
        float e0 = __builtin_amdgcn_exp2f(s0 * L2E - K1);
        float e1 = __builtin_amdgcn_exp2f(s1 * L2E - K1);
        accS[0][v] += (d == v)       ? 0.0f : e0;
        accS[1][v] += (d - 16 == v)  ? 0.0f : e1;
        posS[0][v] += (dp == v)      ? s0 : 0.0f;
        posS[1][v] += (dp - 16 == v) ? s1 : 0.0f;
      }
    } else {
      // Hot path (1020/1024 tiles): no masks, just fma + exp + accumulate.
      #pragma unroll
      for (int v = 0; v < 8; ++v) {
        accS[0][v] += __builtin_amdgcn_exp2f(c0[v] * K1 - K1);
        accS[1][v] += __builtin_amdgcn_exp2f(c1[v] * K1 - K1);
      }
    }
  }

  // Reduce the 16 lanes of each half-wave (xor 1,2,4,8 stays in the group).
  #pragma unroll
  for (int rt = 0; rt < ROWT; ++rt) {
    #pragma unroll
    for (int v = 0; v < 8; ++v) {
      float a = accS[rt][v], p = posS[rt][v];
      #pragma unroll
      for (int off = 1; off < 16; off <<= 1) {
        a += __shfl_xor(a, off, 32);
        p += __shfl_xor(p, off, 32);
      }
      if (ln == 0) {
        // lse = ln2*log2(sum) + 2 ; loss_r = lse - s_pos ; mean over 2B rows
        float lse = 0.6931471805599453f * __builtin_amdgcn_logf(a) + 2.0f;
        atomicAdd(out, (lse - p) * (1.0f / (float)N_ROWS));
      }
    }
  }
}

// ---------------------------------------------------------------------------
extern "C" void kernel_launch(void* const* d_in, const int* in_sizes, int n_in,
                              void* d_out, int out_size, void* d_ws, size_t ws_size,
                              hipStream_t stream) {
  const float* zi = (const float*)d_in[0];
  const float* zj = (const float*)d_in[1];
  float* out = (float*)d_out;
  _Float16* rep = (_Float16*)d_ws;     // 16384 x 128 f16 = 4 MB scratch

  ntx_normalize_kernel<<<N_ROWS / 8, 256, 0, stream>>>(zi, zj, rep);
  ntx_zero_kernel<<<1, 1, 0, stream>>>(out);
  ntx_main_kernel<<<N_ROWS / (16 * ROWT * WAVES_PER_WG), 32 * WAVES_PER_WG, 0,
                    stream>>>(rep, out);
}
